// MultiHeadAttention_20126216749094
// MI455X (gfx1250) — matrix operand order repaired and checked by reading, not compiled, not run on hardware
//
#include <hip/hip_runtime.h>
#include <hip/hip_bf16.h>
#include <math.h>

typedef __attribute__((ext_vector_type(16))) _Float16 v16h;
typedef __attribute__((ext_vector_type(8)))  _Float16 v8h;
typedef __attribute__((ext_vector_type(8)))  float    v8f;
typedef __attribute__((ext_vector_type(4)))  float    v4f;

#define B_   32
#define NQ_  1024
#define GS_  1024
#define ED_  128
#define NH_  8
#define HD_  16

__device__ __forceinline__ v8f wmma_f16(v16h a, v16h b, v8f c) {
  // D = A(16x32 f16) * B(32x16 f16) + C(16x16 f32)
  return __builtin_amdgcn_wmma_f32_16x16x32_f16(false, a, false, b, (short)0, c, false, false);
}

__device__ __forceinline__ float redmax16(float v) {
  v = fmaxf(v, __shfl_xor(v, 1, 32));
  v = fmaxf(v, __shfl_xor(v, 2, 32));
  v = fmaxf(v, __shfl_xor(v, 4, 32));
  v = fmaxf(v, __shfl_xor(v, 8, 32));
  return v;
}
__device__ __forceinline__ float redsum16(float v) {
  v += __shfl_xor(v, 1, 32);
  v += __shfl_xor(v, 2, 32);
  v += __shfl_xor(v, 4, 32);
  v += __shfl_xor(v, 8, 32);
  return v;
}

// ---------------------------------------------------------------------------
// Kernel 0: transpose weights to f16, B-operand friendly layouts.
//   WqT/WkT/WvT : [h][e=16][d=128]   (WT[h][e][d] = W[h][d][e])
//   WoutT       : [d=128][he=128]    (WoutT[d][he] = Wout[he/16][he%16][d])
// ---------------------------------------------------------------------------
__global__ void k_prep(const float* __restrict__ Wq, const float* __restrict__ Wk,
                       const float* __restrict__ Wv, const float* __restrict__ Wout,
                       _Float16* __restrict__ WqT, _Float16* __restrict__ WkT,
                       _Float16* __restrict__ WvT, _Float16* __restrict__ WoutT) {
  int i = blockIdx.x * blockDim.x + threadIdx.x;
  if (i >= NH_ * ED_ * HD_) return;                  // 16384
  int hh  = i / (ED_ * HD_);
  int rem = i % (ED_ * HD_);
  int d   = rem / HD_;
  int e   = rem % HD_;
  int dst = (hh * HD_ + e) * ED_ + d;
  WqT[dst] = (_Float16)Wq[i];
  WkT[dst] = (_Float16)Wk[i];
  WvT[dst] = (_Float16)Wv[i];
  int he = i / ED_;                                  // Wout flat: [he][d2]
  int d2 = i % ED_;
  WoutT[d2 * ED_ + he] = (_Float16)Wout[i];
}

// ---------------------------------------------------------------------------
// Kernel 1: projections.  One block = (batch b, 16-row tile); 8 waves = 8 heads.
//   Qh : f16 [b][h][n][16]   (A / score-B friendly, row contiguous)
//   Kh : f16 [b][h][g][16]
//   VT : f16 [b][h][e=16][g] (attn*V B-operand friendly)
// ---------------------------------------------------------------------------
__global__ void __launch_bounds__(256) k_project(
    const float* __restrict__ q, const float* __restrict__ h,
    const _Float16* __restrict__ WqT, const _Float16* __restrict__ WkT,
    const _Float16* __restrict__ WvT,
    _Float16* __restrict__ Qh, _Float16* __restrict__ Kh, _Float16* __restrict__ VT) {
  const int lane  = threadIdx.x & 31;
  const int head  = threadIdx.x >> 5;
  const int n0    = blockIdx.x * 16;
  const int b     = blockIdx.y;
  const int mn    = lane & 15;
  const int rhalf = (lane < 16) ? 0 : 8;
  const int koffA = (lane < 16) ? 0 : 16; // A: K=0..15 / 16..31
  const int koffB = (lane < 16) ? 0 : 16; // B: K=0..15 / 16..31

  // A tiles (16 x 128 in 4 chunks of K=32), fp32 -> f16
  v16h aq[4], ah[4];
#pragma unroll
  for (int c = 0; c < 4; ++c) {
    const int kb = c * 32;
    const float* qp = q + ((size_t)b * NQ_ + n0 + mn) * ED_ + kb + koffA;
    const float* hp = h + ((size_t)b * GS_ + n0 + mn) * ED_ + kb + koffA;
    v4f q0 = *(const v4f*)(qp);      v4f q1 = *(const v4f*)(qp + 4);
    v4f q2 = *(const v4f*)(qp + 8); v4f q3 = *(const v4f*)(qp + 12);
    v4f h0 = *(const v4f*)(hp);      v4f h1 = *(const v4f*)(hp + 4);
    v4f h2 = *(const v4f*)(hp + 8); v4f h3 = *(const v4f*)(hp + 12);
#pragma unroll
    for (int i = 0; i < 4; ++i) {
      aq[c][i]     = (_Float16)q0[i]; aq[c][i + 4]  = (_Float16)q1[i];
      aq[c][i + 8] = (_Float16)q2[i]; aq[c][i + 12] = (_Float16)q3[i];
      ah[c][i]     = (_Float16)h0[i]; ah[c][i + 4]  = (_Float16)h1[i];
      ah[c][i + 8] = (_Float16)h2[i]; ah[c][i + 12] = (_Float16)h3[i];
    }
  }

  const _Float16* wq = WqT + ((size_t)head * HD_ + mn) * ED_ + koffB;
  const _Float16* wk = WkT + ((size_t)head * HD_ + mn) * ED_ + koffB;
  const _Float16* wv = WvT + ((size_t)head * HD_ + mn) * ED_ + koffB;

  v8f cq = {}, ck = {}, cv = {};
#pragma unroll
  for (int c = 0; c < 4; ++c) {
    cq = wmma_f16(aq[c], *(const v16h*)(wq + c * 32), cq);
    ck = wmma_f16(ah[c], *(const v16h*)(wk + c * 32), ck);
    cv = wmma_f16(ah[c], *(const v16h*)(wv + c * 32), cv);
  }

  const size_t bh = (size_t)b * NH_ + head;
  _Float16* qdst = Qh + (bh * NQ_ + n0) * HD_;
  _Float16* kdst = Kh + (bh * GS_ + n0) * HD_;
#pragma unroll
  for (int r = 0; r < 8; ++r) {
    qdst[(size_t)(r + rhalf) * HD_ + mn] = (_Float16)cq[r];
    kdst[(size_t)(r + rhalf) * HD_ + mn] = (_Float16)ck[r];
    VT[(bh * HD_ + mn) * GS_ + n0 + r + rhalf] = (_Float16)cv[r];
  }
}

// ---------------------------------------------------------------------------
// Kernel 2: fused flash attention.  One wave per (b, head, 16-query tile).
// Loops over 32-key chunks: 2 score WMMAs (K-dim=16, zero-padded to 32),
// online softmax (shfl reductions within 16-lane halves), P via LDS, 1
// accumulate WMMA against V^T.  Writes heads concatenated: hbuf[b][n][h*16+e].
// ---------------------------------------------------------------------------
__global__ void __launch_bounds__(32) k_attention(
    const _Float16* __restrict__ Qh, const _Float16* __restrict__ Kh,
    const _Float16* __restrict__ VT, const unsigned char* __restrict__ mask,
    _Float16* __restrict__ hbuf) {
  __shared__ __align__(64) _Float16 Plds[16 * 32];
  const int lane  = threadIdx.x;
  const int q0    = blockIdx.x * 16;
  const int head  = blockIdx.y;
  const int b     = blockIdx.z;
  const int mn    = lane & 15;
  const int rhalf = (lane < 16) ? 0 : 8;
  const size_t bh = (size_t)b * NH_ + head;

  // A operand: Q tile, contraction = 16 real values, upper 16 zero.
  v16h qa;
#pragma unroll
  for (int i = 0; i < 16; ++i) qa[i] = (_Float16)0.f;
  if (lane < 16) {
    qa = *(const v16h*)(Qh + (bh * NQ_ + q0 + mn) * HD_);
  }

  v8f acc = {};
  float mrun[8], lrun[8];
#pragma unroll
  for (int r = 0; r < 8; ++r) { mrun[r] = -1e30f; lrun[r] = 0.f; }

  const unsigned char* mbase = mask + ((size_t)b * NQ_ + q0) * GS_;
  const int koffB = (lane < 16) ? 0 : 16;

  for (int g0 = 0; g0 < GS_; g0 += 32) {
    // Score B operands: lanes 0-15 hold K[g0+n][0..15]; lanes 16-31 are the
    // zero-padded K=16..31 region.
    v16h kb0, kb1;
#pragma unroll
    for (int i = 0; i < 16; ++i) { kb0[i] = (_Float16)0.f; kb1[i] = (_Float16)0.f; }
    if (lane < 16) {
      kb0 = *(const v16h*)(Kh + (bh * GS_ + g0 + mn) * HD_);
      kb1 = *(const v16h*)(Kh + (bh * GS_ + g0 + 16 + mn) * HD_);
    }
    v8f z = {};
    v8f s0v = wmma_f16(qa, kb0, z);
    v8f s1v = wmma_f16(qa, kb1, z);

    float p0[8], p1[8];
#pragma unroll
    for (int r = 0; r < 8; ++r) {
      float s0 = 0.25f * s0v[r];          // 1/sqrt(HDIM)
      float s1 = 0.25f * s1v[r];
      const unsigned char* mp = mbase + (size_t)(r + rhalf) * GS_ + g0 + mn;
      if (mp[0])  s0 = -1e30f;
      if (mp[16]) s1 = -1e30f;
      float cm    = redmax16(fmaxf(s0, s1));
      float mnew  = fmaxf(mrun[r], cm);
      float alpha = __expf(mrun[r] - mnew);
      p0[r] = __expf(s0 - mnew);
      p1[r] = __expf(s1 - mnew);
      float rs = redsum16(p0[r] + p1[r]);
      lrun[r]  = lrun[r] * alpha + rs;
      acc[r]  *= alpha;
      mrun[r]  = mnew;
    }
    // Stage P (16x32, row-major) so it can be reloaded in A layout.
#pragma unroll
    for (int r = 0; r < 8; ++r) {
      Plds[(r + rhalf) * 32 + mn]      = (_Float16)p0[r];
      Plds[(r + rhalf) * 32 + 16 + mn] = (_Float16)p1[r];
    }
    __syncthreads();
    v16h pa;
    {
      const _Float16* pp = &Plds[mn * 32 + koffB];
      v8h lo = *(const v8h*)pp;
      v8h hi = *(const v8h*)(pp + 8);
#pragma unroll
      for (int i = 0; i < 8; ++i) { pa[i] = lo[i]; pa[i + 8] = hi[i]; }
    }
    // V B-operand: column e=mn, K=g0+koffB..+15 contiguous in V^T.
    v16h vb = *(const v16h*)(VT + (bh * HD_ + mn) * GS_ + g0 + koffB);
    acc = wmma_f16(pa, vb, acc);
    __syncthreads();
  }

  _Float16* dst = hbuf + ((size_t)b * NQ_ + q0) * ED_ + head * HD_;
#pragma unroll
  for (int r = 0; r < 8; ++r)
    dst[(size_t)(r + rhalf) * ED_ + mn] = (_Float16)(acc[r] / lrun[r]);
}

// ---------------------------------------------------------------------------
// Kernel 3: output projection: out[b][n][d] = hbuf[b][n][:] dot WoutT[d][:].
// One block = (b, 16-row tile); wave w handles columns d0 = w*16.
// ---------------------------------------------------------------------------
__global__ void __launch_bounds__(256) k_outproj(
    const _Float16* __restrict__ hbuf, const _Float16* __restrict__ WoutT,
    float* __restrict__ out) {
  const int lane  = threadIdx.x & 31;
  const int d0    = (threadIdx.x >> 5) * 16;
  const int n0    = blockIdx.x * 16;
  const int b     = blockIdx.y;
  const int mn    = lane & 15;
  const int rhalf = (lane < 16) ? 0 : 8;
  const int koffA = rhalf;
  const int koffB = (lane < 16) ? 0 : 16;

  v8f c = {};
  const _Float16* arow = hbuf + ((size_t)b * NQ_ + n0 + mn) * ED_;
  const _Float16* brow = WoutT + (size_t)(d0 + mn) * ED_;
#pragma unroll
  for (int ck = 0; ck < 4; ++ck) {
    const int kb = ck * 32;
    v8h lo = *(const v8h*)(arow + kb + koffB);
    v8h hi = *(const v8h*)(arow + kb + koffB + 8);
    v16h a;
#pragma unroll
    for (int i = 0; i < 8; ++i) { a[i] = lo[i]; a[i + 8] = hi[i]; }
    v16h bb = *(const v16h*)(brow + kb + koffB);
    c = wmma_f16(a, bb, c);
  }
  float* dst = out + ((size_t)b * NQ_ + n0) * ED_ + d0;
#pragma unroll
  for (int r = 0; r < 8; ++r)
    dst[(size_t)(r + rhalf) * ED_ + mn] = c[r];   // lanes 0..15 coalesced
}

// ---------------------------------------------------------------------------
extern "C" void kernel_launch(void* const* d_in, const int* in_sizes, int n_in,
                              void* d_out, int out_size, void* d_ws, size_t ws_size,
                              hipStream_t stream) {
  (void)in_sizes; (void)n_in; (void)out_size; (void)ws_size;
  const float*         q    = (const float*)d_in[0];
  const float*         h    = (const float*)d_in[1];
  const unsigned char* mask = (const unsigned char*)d_in[2];  // jax bool = 1 byte
  const float*         Wq   = (const float*)d_in[3];
  const float*         Wk   = (const float*)d_in[4];
  const float*         Wv   = (const float*)d_in[5];
  const float*         Wout = (const float*)d_in[6];
  float*               out  = (float*)d_out;

  char* ws = (char*)d_ws;
  const size_t wtsz  = (size_t)NH_ * HD_ * ED_ * sizeof(_Float16);   // 32 KB
  const size_t qkvsz = (size_t)B_ * NH_ * NQ_ * HD_ * sizeof(_Float16); // 8 MB
  _Float16* WqT   = (_Float16*)(ws);                 ws += wtsz;
  _Float16* WkT   = (_Float16*)(ws);                 ws += wtsz;
  _Float16* WvT   = (_Float16*)(ws);                 ws += wtsz;
  _Float16* WoutT = (_Float16*)(ws);                 ws += wtsz;
  _Float16* Qh    = (_Float16*)(ws);                 ws += qkvsz;
  _Float16* Kh    = (_Float16*)(ws);                 ws += qkvsz;
  _Float16* VT    = (_Float16*)(ws);                 ws += qkvsz;
  _Float16* hbuf  = (_Float16*)(ws);

  k_prep<<<dim3((NH_ * ED_ * HD_ + 255) / 256), dim3(256), 0, stream>>>(
      Wq, Wk, Wv, Wout, WqT, WkT, WvT, WoutT);
  k_project<<<dim3(NQ_ / 16, B_), dim3(256), 0, stream>>>(
      q, h, WqT, WkT, WvT, Qh, Kh, VT);
  k_attention<<<dim3(NQ_ / 16, NH_, B_), dim3(32), 0, stream>>>(
      Qh, Kh, VT, mask, hbuf);
  k_outproj<<<dim3(NQ_ / 16, B_), dim3(256), 0, stream>>>(hbuf, WoutT, out);
}
